// MultiHeadSelfAttention_70102456205580
// MI455X (gfx1250) — matrix operand order repaired and checked by reading, not compiled, not run on hardware
//
#include <hip/hip_runtime.h>
#include <hip/hip_bf16.h>

// ---------------------------------------------------------------------------
// MI455X (gfx1250) multi-head self-attention, bf16 WMMA with f32 accumulation.
// Pipeline: cvt(x,W*) -> GEMM Q,K,V (V transposed per-head) -> flash attention
//           (K*Q^T scores-transposed trick) -> GEMM output projection (f32 out)
// Round 2 additions: software pipelining. GEMM: 2-stage ping-pong operand
// buffers (K unrolled by 64). Flash: K A-operands double-buffered across
// chunks; V loads issued before the softmax VALU block so exp/rescale work
// hides their latency. All pipeline control is scalar-uniform (EXEC all-ones
// around every WMMA).
// ---------------------------------------------------------------------------

typedef __bf16 bf16_t;
typedef __attribute__((ext_vector_type(16))) __bf16       v16bf;
typedef __attribute__((ext_vector_type(8)))  float        v8f;
typedef __attribute__((ext_vector_type(4)))  unsigned int u32x4;
typedef __attribute__((ext_vector_type(4)))  __bf16       v4bf;

#define DMODEL 1024
#define SEQ    2048
#define BATCH  4
#define HEADS  16
#define DK     64
#define ROWS   (BATCH * SEQ)          // 8192

__device__ __forceinline__ int rfl(int x) {
  return __builtin_amdgcn_readfirstlane(x);
}

// ---------------- WMMA helpers (layouts per CDNA5 ISA 7.12.2) ---------------

__device__ __forceinline__ v8f wmma_bf16(v16bf a, v16bf b, v8f c) {
  // D = A(16x32 bf16) * B(32x16 bf16) + C(16x16 f32)
  return __builtin_amdgcn_wmma_f32_16x16x32_bf16(
      /*neg_a=*/false, a, /*neg_b=*/false, b,
      /*c_mod=*/(short)0, c, /*reuse_a=*/false, /*reuse_b=*/false);
}

// A-operand: lane L (m = L&15, hi = L>>4) holds
//   a[0..7]  = A[m][hi*8 + 0..7]
//   a[8..15] = A[m][16 + hi*8 + 0..7]
__device__ __forceinline__ v16bf load_a16(const bf16_t* p) {
  union { v16bf v; u32x4 q[2]; } u;
  u.q[0] = *(const u32x4*)(p);
  u.q[1] = *(const u32x4*)(p + 16);
  return u.v;
}

// B-operand: lane L (n = L&15, hi = L>>4) holds
//   b[0..7] = B[hi*8 + 0..7][n], b[8..15] = B[16 + hi*8 + 0..7][n].
__device__ __forceinline__ v16bf load_b16(const bf16_t* p) {
  union { v16bf v; u32x4 q[2]; } u;
  u.q[0] = *(const u32x4*)(p);
  u.q[1] = *(const u32x4*)(p + 16);
  return u.v;
}

// ---------------------------- f32 -> bf16 convert ---------------------------

__global__ __launch_bounds__(256) void cvt_f32_bf16(
    const float* __restrict__ in, bf16_t* __restrict__ out, int n4) {
  int i = blockIdx.x * blockDim.x + threadIdx.x;
  if (i >= n4) return;
  float4 f = ((const float4*)in)[i];
  v4bf o;
  o[0] = (bf16_t)f.x; o[1] = (bf16_t)f.y; o[2] = (bf16_t)f.z; o[3] = (bf16_t)f.w;
  ((v4bf*)out)[i] = o;
}

// ------------------------------- GEMM ---------------------------------------
// C[M,N] = A[M,K] * W[N,K]^T     (Linear with weight stored (d_out, d_in))
// One wave computes a 32x64 block: 2 M-tiles x 4 N-tiles. K stepped by 32,
// software-pipelined 2 stages (K must be a multiple of 64).
// MODE 0: bf16 row-major.  MODE 1: bf16 as V^T [b,h,dk,s].  MODE 2: f32.

template <int MODE>
__global__ __launch_bounds__(256) void gemm_bf16(
    const bf16_t* __restrict__ A, const bf16_t* __restrict__ W,
    void* __restrict__ Cout, int M, int N, int K) {
  const int lane = threadIdx.x & 31;
  const int wid  = rfl(blockIdx.x * 8 + (threadIdx.x >> 5));  // wave-uniform
  const int nBlk = N >> 6;              // 64-wide N blocks
  const int m0   = (wid / nBlk) * 32;   // SGPR
  const int n0   = (wid % nBlk) * 64;   // SGPR
  const int li   = lane & 15, hi = lane >> 4;

  v8f acc[2][4];
#pragma unroll
  for (int mt = 0; mt < 2; ++mt)
#pragma unroll
    for (int nt = 0; nt < 4; ++nt)
#pragma unroll
      for (int j = 0; j < 8; ++j) acc[mt][nt][j] = 0.0f;

  const bf16_t* a0 = A + (size_t)(m0 + li) * K + hi * 8;
  const bf16_t* a1 = A + (size_t)(m0 + 16 + li) * K + hi * 8;
  const bf16_t* w0 = W + (size_t)(n0 + 0  + li) * K + hi * 8;
  const bf16_t* w1 = W + (size_t)(n0 + 16 + li) * K + hi * 8;
  const bf16_t* w2 = W + (size_t)(n0 + 32 + li) * K + hi * 8;
  const bf16_t* w3 = W + (size_t)(n0 + 48 + li) * K + hi * 8;

  v16bf opA[2][6], opB[2][6];  // ping-pong operand stages
  auto loadStage = [&](v16bf(&st)[6], int k0) {
    st[0] = load_a16(a0 + k0);
    st[1] = load_a16(a1 + k0);
    st[2] = load_b16(w0 + k0);
    st[3] = load_b16(w1 + k0);
    st[4] = load_b16(w2 + k0);
    st[5] = load_b16(w3 + k0);
  };
  auto mmaStage = [&](v16bf(&st)[6]) {
#pragma unroll
    for (int nt = 0; nt < 4; ++nt) {
      acc[0][nt] = wmma_bf16(st[0], st[2 + nt], acc[0][nt]);
      acc[1][nt] = wmma_bf16(st[1], st[2 + nt], acc[1][nt]);
    }
  };

  loadStage(opA[0], 0);
  for (int k0 = 0; k0 < K; k0 += 64) {
    loadStage(opA[1], k0 + 32);       // stage B loads fly over stage A MMAs
    mmaStage(opA[0]);
    if (k0 + 64 < K) loadStage(opA[0], k0 + 64);
    mmaStage(opA[1]);
  }
  (void)opB;

#pragma unroll
  for (int mt = 0; mt < 2; ++mt) {
#pragma unroll
    for (int nt = 0; nt < 4; ++nt) {
#pragma unroll
      for (int j = 0; j < 8; ++j) {
        const int row = m0 + mt * 16 + hi * 8 + j;   // C layout: M = hi*8+j
        const int col = n0 + nt * 16 + li;           //           N = lane&15
        const float v = acc[mt][nt][j];
        if (MODE == 0) {
          ((bf16_t*)Cout)[(size_t)row * N + col] = (bf16_t)v;
        } else if (MODE == 1) {
          // V^T: row=(b,s), col=(h,d) -> Vt[((b*H+h)*DK+d)*SEQ + s]
          const int b = row >> 11, s = row & (SEQ - 1);
          const int h = col >> 6,  d = col & (DK - 1);
          ((bf16_t*)Cout)[(((size_t)(b * HEADS + h) * DK + d) * SEQ) + s] =
              (bf16_t)v;
        } else {
          ((float*)Cout)[(size_t)row * N + col] = v;
        }
      }
    }
  }
}

// --------------------------- flash attention --------------------------------
// One wave handles one (b, h, 16-query tile). Scores computed TRANSPOSED
// (K * Q^T) so each lane owns one query column -> softmax reductions are
// 8 register maxes + one shfl_xor(16); the exp'd score registers directly
// form the A-operand of the P*V WMMA. Exactly one chunk (the diagonal one)
// needs causal masking. K operands are double-buffered across chunks; V
// loads are issued before the softmax VALU block.

__global__ __launch_bounds__(256) void flash_attn(
    const bf16_t* __restrict__ Qm, const bf16_t* __restrict__ Km,
    const bf16_t* __restrict__ Vt, bf16_t* __restrict__ Om) {
  const int lane = threadIdx.x & 31;
  const int task = rfl(blockIdx.x * 8 + (threadIdx.x >> 5));  // wave-uniform
  const int qt   = task & 127;                   // query tile within sequence
  const int bh   = task >> 7;                    // (b*H + h)
  const int q0   = qt * 16;                      // SGPR
  const int b    = bh >> 4;
  const int h    = bh & 15;
  const int li   = lane & 15, hi = lane >> 4;

  const bf16_t* Qh = Qm + (size_t)b * SEQ * DMODEL + h * DK;
  const bf16_t* Kh = Km + (size_t)b * SEQ * DMODEL + h * DK;
  const bf16_t* Vh = Vt + (size_t)bh * DK * SEQ;

  // Preload Q^T B-operands: bq[ks] covers d = ks*32 .. ks*32+31
  v16bf bq[2];
#pragma unroll
  for (int ks = 0; ks < 2; ++ks)
    bq[ks] = load_b16(Qh + (size_t)(q0 + li) * DMODEL + ks * 32 + hi * 8);

  v8f o[4];
#pragma unroll
  for (int dt = 0; dt < 4; ++dt)
#pragma unroll
    for (int j = 0; j < 8; ++j) o[dt][j] = 0.0f;

  float Mx = -1e30f;   // running row max (query = lane&15)
  float Sm = 0.0f;     // running partial exp-sum (this lane-half's keys)
  const float CE = 0.18033688011112042f;  // (1/sqrt(dk)) * log2(e)
  const int q = q0 + li;
  const int mDiag = (q0 >> 5) * 32;  // the single chunk needing causal mask

  auto loadK = [&](int m0, v16bf(&kk)[4]) {
    const bf16_t* kr0 = Kh + (size_t)(m0 + li) * DMODEL + hi * 8;
    const bf16_t* kr1 = Kh + (size_t)(m0 + 16 + li) * DMODEL + hi * 8;
    kk[0] = load_a16(kr0);
    kk[1] = load_a16(kr0 + 32);
    kk[2] = load_a16(kr1);
    kk[3] = load_a16(kr1 + 32);
  };

  // chunk body: consumes kc (current chunk), prefetches into kn (next chunk)
  auto body = [&](v16bf(&kc)[4], v16bf(&kn)[4], int m0) -> bool {
    v8f s0, s1;
#pragma unroll
    for (int j = 0; j < 8; ++j) { s0[j] = 0.0f; s1[j] = 0.0f; }
    s0 = wmma_bf16(kc[0], bq[0], s0);
    s0 = wmma_bf16(kc[1], bq[1], s0);
    s1 = wmma_bf16(kc[2], bq[0], s1);
    s1 = wmma_bf16(kc[3], bq[1], s1);

    const bool hasNext = (m0 + 32 <= mDiag);   // scalar-uniform
    if (hasNext) loadK(m0 + 32, kn);           // prefetch next chunk's K

    v16bf vv[4];                               // V loads fly over softmax VALU
#pragma unroll
    for (int dt = 0; dt < 4; ++dt)
      vv[dt] = load_b16(Vh + (size_t)(dt * 16 + li) * SEQ + m0 + hi * 8);

    if (m0 == mDiag) {                         // peeled causal mask (uniform)
#pragma unroll
      for (int j = 0; j < 8; ++j) {
        const int key = m0 + hi * 8 + j;
        s0[j] = (key > q)      ? -1e30f : s0[j];
        s1[j] = (key + 16 > q) ? -1e30f : s1[j];
      }
    }

    // online softmax along keys (register dim + opposite lane half)
    float cm = -1e30f;
#pragma unroll
    for (int j = 0; j < 8; ++j) cm = fmaxf(cm, fmaxf(s0[j], s1[j]));
    cm = fmaxf(cm, __shfl_xor(cm, 16, 32));
    const float newM = fmaxf(Mx, cm);
    const float so   = __builtin_exp2f((Mx - newM) * CE);
    Mx = newM;

    v16bf pa;          // exp'd scores -> A operand of P*V (layout matches!)
    float ps = 0.0f;
#pragma unroll
    for (int j = 0; j < 8; ++j) {
      const float p0 = __builtin_exp2f((s0[j] - newM) * CE);
      const float p1 = __builtin_exp2f((s1[j] - newM) * CE);
      ps += p0 + p1;
      pa[j]     = (bf16_t)p0;
      pa[8 + j] = (bf16_t)p1;
    }
    Sm = Sm * so + ps;

    // rescale output accumulators: broadcast per-query factor to out lanes
    float sf[8];
#pragma unroll
    for (int j = 0; j < 8; ++j) sf[j] = __shfl(so, hi * 8 + j, 32);
#pragma unroll
    for (int dt = 0; dt < 4; ++dt)
#pragma unroll
      for (int j = 0; j < 8; ++j) o[dt][j] *= sf[j];

    // P(16x32) * V(32x64): 4 d-tiles
#pragma unroll
    for (int dt = 0; dt < 4; ++dt) o[dt] = wmma_bf16(pa, vv[dt], o[dt]);

    return hasNext;
  };

  // 2x-peeled ping-pong so buffer indices stay compile-time constants
  v16bf kA[4], kB[4];
  loadK(0, kA);
  int m0 = 0;
  for (;;) {
    if (!body(kA, kB, m0)) break;
    m0 += 32;
    if (!body(kB, kA, m0)) break;
    m0 += 32;
  }

  // finalize: combine lane-half sums, divide, store bf16 context
  const float St  = Sm + __shfl_xor(Sm, 16, 32);
  const float inv = 1.0f / St;
  float rf[8];
#pragma unroll
  for (int j = 0; j < 8; ++j) rf[j] = __shfl(inv, hi * 8 + j, 32);

  bf16_t* Orow = Om + ((size_t)b * SEQ + q0) * DMODEL + h * DK;
#pragma unroll
  for (int dt = 0; dt < 4; ++dt)
#pragma unroll
    for (int j = 0; j < 8; ++j)
      Orow[(size_t)(hi * 8 + j) * DMODEL + dt * 16 + li] =
          (bf16_t)(o[dt][j] * rf[j]);
}

// ------------------------------- launcher -----------------------------------

extern "C" void kernel_launch(void* const* d_in, const int* in_sizes, int n_in,
                              void* d_out, int out_size, void* d_ws,
                              size_t ws_size, hipStream_t stream) {
  const float* x  = (const float*)d_in[0];
  const float* Wq = (const float*)d_in[1];
  const float* Wk = (const float*)d_in[2];
  const float* Wv = (const float*)d_in[3];
  const float* Wo = (const float*)d_in[4];

  // workspace layout (bf16 elements)
  const size_t NX = (size_t)ROWS * DMODEL;     // 8M
  const size_t NW = (size_t)DMODEL * DMODEL;   // 1M
  bf16_t* xb = (bf16_t*)d_ws;
  bf16_t* wq = xb + NX;
  bf16_t* wk = wq + NW;
  bf16_t* wv = wk + NW;
  bf16_t* wo = wv + NW;
  bf16_t* Qb = wo + NW;
  bf16_t* Kb = Qb + NX;
  bf16_t* Vt = Kb + NX;   // [b,h,dk,s]
  bf16_t* Ab = Vt + NX;   // attention context, bf16 row-major

  // f32 -> bf16 converts (vectorized x4)
  cvt_f32_bf16<<<(int)(NX / 4 / 256), 256, 0, stream>>>(x, xb, (int)(NX / 4));
  cvt_f32_bf16<<<(int)(NW / 4 / 256), 256, 0, stream>>>(Wq, wq, (int)(NW / 4));
  cvt_f32_bf16<<<(int)(NW / 4 / 256), 256, 0, stream>>>(Wk, wk, (int)(NW / 4));
  cvt_f32_bf16<<<(int)(NW / 4 / 256), 256, 0, stream>>>(Wv, wv, (int)(NW / 4));
  cvt_f32_bf16<<<(int)(NW / 4 / 256), 256, 0, stream>>>(Wo, wo, (int)(NW / 4));

  // projections: (8192/32)*(1024/64) = 4096 waves -> 512 blocks of 8 waves
  gemm_bf16<0><<<512, 256, 0, stream>>>(xb, wq, Qb, ROWS, DMODEL, DMODEL);
  gemm_bf16<0><<<512, 256, 0, stream>>>(xb, wk, Kb, ROWS, DMODEL, DMODEL);
  gemm_bf16<1><<<512, 256, 0, stream>>>(xb, wv, Vt, ROWS, DMODEL, DMODEL);

  // flash attention: 8192 wave-tasks -> 1024 blocks of 8 waves
  flash_attn<<<1024, 256, 0, stream>>>(Qb, Kb, Vt, Ab);

  // output projection, f32 straight to d_out
  gemm_bf16<2><<<512, 256, 0, stream>>>(Ab, wo, d_out, ROWS, DMODEL, DMODEL);
}
